// SceneGraphEncoder_2963527434558
// MI455X (gfx1250) — matrix operand order repaired and checked by reading, not compiled, not run on hardware
//
#include <hip/hip_runtime.h>

#define Bn 8
#define Nn 1024
#define En 8192
#define Dn 256

typedef _Float16 h16_t;
typedef _Float16 v16h __attribute__((ext_vector_type(16)));
typedef _Float16 v8h  __attribute__((ext_vector_type(8)));
typedef _Float16 v4h  __attribute__((ext_vector_type(4)));
typedef float    v8f  __attribute__((ext_vector_type(8)));
typedef int      v2i  __attribute__((ext_vector_type(2)));

// pointer types for the async direct-to-LDS builtin (b64 payload = int2)
typedef __attribute__((address_space(1))) v2i* g_v2i_p;
typedef __attribute__((address_space(3))) v2i* l_v2i_p;

#if defined(__HIP_DEVICE_COMPILE__) && defined(__has_builtin)
#if __has_builtin(__builtin_amdgcn_global_load_async_to_lds_b64)
#define HAVE_ASYNC_LDS 1
#endif
#endif
#ifndef HAVE_ASYNC_LDS
#define HAVE_ASYNC_LDS 0
#endif

enum { LD_F16 = 0, LD_F32, LD_GATHER_F32, LD_CONCAT2_F16, LD_TRIPLET };
enum { EP_F16 = 0, EP_SILU_F16, EP_FUSE, EP_MSG, EP_EDGE_RES, EP_NODE_LN };

struct GemmP {
  const void* A;            // primary A source (f16 or f32 depending on mode)
  const void* A2;           // second concat source / edge_feat for triplet
  const int* eidx;          // gather indices (node types) or edge_index (B,2,E)
  const h16_t* W;           // packed weights, N-major: W[n*K + k]
  const float* bias;        // per-column bias (always present)
  const unsigned char* mask;// per-row mask (nm or em)
  float* outF32;            // f32 output / residual target (node_feat)
  h16_t* outF16;            // f16 output (or in/out for edge residual)
  float* agg;               // scatter target for msg epilogue
  const float* lng;         // layernorm gain
  const float* lnb;         // layernorm bias
  int M;                    // rows (multiple of 16)
  int K;                    // reduction dim (multiple of 32)
};

__device__ __forceinline__ float silu_f(float x) { return x / (1.0f + __expf(-x)); }

__device__ __forceinline__ void wait_async0() {
#if defined(__has_builtin) && __has_builtin(__builtin_amdgcn_s_wait_asynccnt)
  __builtin_amdgcn_s_wait_asynccnt(0);
#else
  asm volatile("s_wait_asynccnt 0x0" ::: "memory");
#endif
}

// ---- A-tile staging: 16x32 f16 per buffer, 128 threads x 4 halves ----
// issueA: start the loads for step k0 (async direct-to-LDS for f16 sources,
// or global->VGPR payload for f32 sources). commitA: convert/store payload.
template <int AM, bool ASY>
__device__ __forceinline__ void issueA(const GemmP& p, const int* rowSrcS,
                                       const int* rowDstS, int rowBase, int k0,
                                       h16_t* dst, float fpay[4], v4h* hpay) {
  const int t = threadIdx.x;
  const int r = t >> 3;            // 0..15 row in tile
  const int kc = (t & 7) << 2;     // 4 halves per thread
  const int row = rowBase + r;
  const int k = k0 + kc;

  if constexpr (AM == LD_F32) {
    const float* s = (const float*)p.A + (size_t)row * p.K + k;
#pragma unroll
    for (int j = 0; j < 4; ++j) fpay[j] = s[j];
  } else if constexpr (AM == LD_GATHER_F32) {
    const float* s = (const float*)p.A + (size_t)rowSrcS[r] * p.K + k;
#pragma unroll
    for (int j = 0; j < 4; ++j) fpay[j] = s[j];
  } else {
    const h16_t* s;
    if constexpr (AM == LD_F16) {
      s = (const h16_t*)p.A + (size_t)row * p.K + k;
    } else if constexpr (AM == LD_CONCAT2_F16) {
      s = (k < Dn) ? (const h16_t*)p.A  + ((size_t)row << 8) + k
                   : (const h16_t*)p.A2 + ((size_t)row << 8) + (k - Dn);
    } else {  // LD_TRIPLET: [src_feat | edge_feat | dst_feat]
      if (k < Dn)          s = (const h16_t*)p.A  + ((size_t)rowSrcS[r] << 8) + k;
      else if (k < 2 * Dn) s = (const h16_t*)p.A2 + ((size_t)row << 8) + (k - Dn);
      else                 s = (const h16_t*)p.A  + ((size_t)rowDstS[r] << 8) + (k - 2 * Dn);
    }
#if HAVE_ASYNC_LDS
    if constexpr (ASY) {
      __builtin_amdgcn_global_load_async_to_lds_b64((g_v2i_p)s, (l_v2i_p)dst, 0, 0);
    } else
#endif
    {
      *hpay = *(const v4h*)s;
    }
  }
  (void)dst;
}

template <int AM, bool ASY>
__device__ __forceinline__ void commitA(h16_t* dst, const float fpay[4],
                                        const v4h* hpay) {
  if constexpr (AM == LD_F32 || AM == LD_GATHER_F32) {
#pragma unroll
    for (int j = 0; j < 4; ++j) dst[j] = (h16_t)fpay[j];
  } else if constexpr (!ASY) {
    *(v4h*)dst = *hpay;
  }
}

// 16-row x 256-col output tile per workgroup; 4 waves, each wave 4 WMMA tiles.
// Software-pipelined: ping-pong LDS A buffers, next stage issued before compute.
template <int AM, int EM>
__global__ __launch_bounds__(128) void gemm_k(GemmP p) {
  constexpr bool ASY =
      HAVE_ASYNC_LDS && (AM == LD_F16 || AM == LD_CONCAT2_F16 || AM == LD_TRIPLET);

  __shared__ __align__(32) h16_t As[2 * 16 * 32];
  __shared__ float Ys[(EM == EP_NODE_LN) ? 16 * Dn : 1];
  __shared__ float muS[16];
  __shared__ float rsS[16];
  __shared__ int rowSrcS[16];
  __shared__ int rowDstS[16];

  const int rowBase = blockIdx.x << 4;
  const int lane = threadIdx.x & 31;
  const int wave = threadIdx.x >> 5;
  const int hs = lane >> 4;
  const int lm = lane & 15;

  // hoist per-row gather indices into LDS (loaded once, reused every K step)
  if constexpr (AM == LD_GATHER_F32) {
    if (threadIdx.x < 16) rowSrcS[threadIdx.x] = p.eidx[rowBase + threadIdx.x];
  } else if constexpr (AM == LD_TRIPLET) {
    if (threadIdx.x < 16) {
      const int row = rowBase + threadIdx.x;
      const int b = row >> 13, e = row & (En - 1);
      rowSrcS[threadIdx.x] = (b << 10) + p.eidx[(size_t)b * 2 * En + e];
      rowDstS[threadIdx.x] = (b << 10) + p.eidx[(size_t)b * 2 * En + En + e];
    }
  }
  if constexpr (AM == LD_GATHER_F32 || AM == LD_TRIPLET) __syncthreads();

  // per-thread LDS staging slot
  const int slot = ((threadIdx.x >> 3) << 5) + ((threadIdx.x & 7) << 2);

  v8f acc[4];
#pragma unroll
  for (int t = 0; t < 4; ++t) acc[t] = (v8f)(0.0f);

  const int nsteps = p.K >> 5;
  float fpay[4];
  v4h hpay;

  // prologue: stage step 0 into buffer 0
  issueA<AM, ASY>(p, rowSrcS, rowDstS, rowBase, 0, &As[slot], fpay, &hpay);
  commitA<AM, ASY>(&As[slot], fpay, &hpay);

  for (int ks = 0; ks < nsteps; ++ks) {
    h16_t* cur = &As[(ks & 1) * (16 * 32)];
    h16_t* nxt = &As[((ks & 1) ^ 1) * (16 * 32)];
    if constexpr (ASY) wait_async0();
    __syncthreads();

    const bool have_next = (ks + 1) < nsteps;
    if (have_next)
      issueA<AM, ASY>(p, rowSrcS, rowDstS, rowBase, (ks + 1) << 5, nxt + slot,
                      fpay, &hpay);

    // A fragment: lanes 0-15 -> K {0..15}; lanes 16-31 -> K {16..31}
    v8h alo = *(const v8h*)&cur[(lm << 5) + (hs << 4)];
    v8h ahi = *(const v8h*)&cur[(lm << 5) + (hs << 4) + 8];
    v16h a;
#pragma unroll
    for (int j = 0; j < 8; ++j) { a[j] = alo[j]; a[j + 8] = ahi[j]; }

    const int k0 = ks << 5;
#pragma unroll
    for (int t = 0; t < 4; ++t) {
      const int ncol = (wave << 6) + (t << 4) + lm;
      // B fragment: contiguous 32B from N-major packed weights (lane=N, elems=K)
      v16h b = *(const v16h*)(p.W + (size_t)ncol * p.K + k0 + (hs << 4));
      acc[t] = __builtin_amdgcn_wmma_f32_16x16x32_f16(
          false, a, false, b, (short)0, acc[t], false, false);
    }

    if (have_next) commitA<AM, ASY>(nxt + slot, fpay, &hpay);
  }

  // ---------------- epilogue ----------------
  const int colBase = wave << 6;
#pragma unroll
  for (int rr = 0; rr < 8; ++rr) {
    const int m = rr + (hs << 3);
    const int row = rowBase + m;
    float mk = 1.0f;
    if constexpr (EM == EP_FUSE || EM == EP_MSG || EM == EP_NODE_LN)
      mk = p.mask[row] ? 1.0f : 0.0f;
    int bb = 0, dn = 0;
    if constexpr (EM == EP_MSG) {
      bb = row >> 13;
      const int e = row & (En - 1);
      dn = p.eidx[(size_t)bb * 2 * En + En + e];
    }
#pragma unroll
    for (int t = 0; t < 4; ++t) {
      const int col = colBase + (t << 4) + lm;
      const float x = acc[t][rr] + p.bias[col];
      const size_t gi = ((size_t)row << 8) + col;
      if constexpr (EM == EP_F16) {
        p.outF16[gi] = (h16_t)x;
      } else if constexpr (EM == EP_SILU_F16) {
        p.outF16[gi] = (h16_t)silu_f(x);
      } else if constexpr (EM == EP_FUSE) {
        const float v = x * mk;
        p.outF32[gi] = v;
        p.outF16[gi] = (h16_t)v;
      } else if constexpr (EM == EP_MSG) {
        const float v = x * mk;
        p.outF16[gi] = (h16_t)v;
        atomicAdd(p.agg + (((size_t)(bb << 10) + dn) << 8) + col, v);
      } else if constexpr (EM == EP_EDGE_RES) {
        p.outF16[gi] = (h16_t)((float)p.outF16[gi] + x);
      } else {  // EP_NODE_LN: residual + mask, stage for layernorm
        Ys[(m << 8) + col] = p.outF32[gi] + x * mk;
      }
    }
  }

  if constexpr (EM == EP_NODE_LN) {
    __syncthreads();
    if (threadIdx.x < 16) {
      float s = 0.0f, s2 = 0.0f;
      const float* yr = &Ys[threadIdx.x << 8];
      for (int j = 0; j < Dn; ++j) { const float v = yr[j]; s += v; s2 += v * v; }
      const float mean = s * (1.0f / Dn);
      const float var = s2 * (1.0f / Dn) - mean * mean;
      muS[threadIdx.x] = mean;
      rsS[threadIdx.x] = rsqrtf(var + 1e-5f);
    }
    __syncthreads();
#pragma unroll 4
    for (int j = 0; j < 32; ++j) {
      const int i = threadIdx.x + (j << 7);
      const int m = i >> 8, col = i & 255;
      const float v = (Ys[i] - muS[m]) * rsS[m] * p.lng[col] + p.lnb[col];
      const size_t gi = (((size_t)rowBase + m) << 8) + col;
      p.outF32[gi] = v;
      p.outF16[gi] = (h16_t)v;
    }
  }
}

// ---- small kernels ----

// Pack weights (K x 256 f32, row-major) -> N-major f16: Wt[n*K + k] = W[k*256 + n]
__global__ __launch_bounds__(256) void pack_k(const float* __restrict__ W,
                                              h16_t* __restrict__ Wt, int K) {
  const int i = blockIdx.x * 256 + threadIdx.x;
  if (i >= K * Dn) return;
  const int n = i / K, k = i - n * K;
  Wt[i] = (h16_t)W[(size_t)k * Dn + n];
}

// geom hidden: K=9 MLP first stage, silu, f16 out
__global__ __launch_bounds__(256) void geom_hidden_k(
    const float* __restrict__ pos, const float* __restrict__ rot,
    const float* __restrict__ siz, const float* __restrict__ W1,
    const float* __restrict__ b1, h16_t* __restrict__ h) {
  const int i = blockIdx.x, j = threadIdx.x;
  float g[9];
#pragma unroll
  for (int q = 0; q < 3; ++q) {
    g[q] = pos[i * 3 + q]; g[3 + q] = rot[i * 3 + q]; g[6 + q] = siz[i * 3 + q];
  }
  float acc = b1[j];
#pragma unroll
  for (int q = 0; q < 9; ++q) acc += g[q] * W1[q * Dn + j];
  h[((size_t)i << 8) + j] = (h16_t)silu_f(acc);
}

__global__ __launch_bounds__(256) void edge_init_k(
    const int* __restrict__ et, const unsigned char* __restrict__ em,
    const float* __restrict__ embed, h16_t* __restrict__ ef) {
  const int i = blockIdx.x * 256 + threadIdx.x;  // over B*E*D
  const int r = i >> 8, c = i & 255;
  const float v = embed[((size_t)(et[r] + 1) << 8) + c] * (em[r] ? 1.0f : 0.0f);
  ef[i] = (h16_t)v;
}

__global__ __launch_bounds__(256) void counts_k(const int* __restrict__ eidx,
                                                const unsigned char* __restrict__ em,
                                                float* __restrict__ counts) {
  const int r = blockIdx.x * 256 + threadIdx.x;  // over B*E
  const int b = r >> 13, e = r & (En - 1);
  if (em[r]) atomicAdd(&counts[(b << 10) + eidx[(size_t)b * 2 * En + En + e]], 1.0f);
}

__global__ __launch_bounds__(256) void aggnorm_k(float* __restrict__ agg,
                                                 const float* __restrict__ counts) {
  const size_t i = (size_t)blockIdx.x * 256 + threadIdx.x;  // over B*N*D
  agg[i] /= fmaxf(counts[i >> 8], 1.0f);
}

__global__ __launch_bounds__(256) void gemb_k(const float* __restrict__ nf,
                                              const unsigned char* __restrict__ nm,
                                              float* __restrict__ out) {
  const int b = blockIdx.x, j = threadIdx.x;
  float s = 0.0f, cnt = 0.0f;
  for (int n = 0; n < Nn; ++n) {
    const float m = nm[(b << 10) + n] ? 1.0f : 0.0f;
    s += nf[(((size_t)(b << 10) + n) << 8) + j] * m;
    cnt += m;
  }
  out[(b << 8) + j] = s / fmaxf(cnt, 1.0f);
}

extern "C" void kernel_launch(void* const* d_in, const int* in_sizes, int n_in,
                              void* d_out, int out_size, void* d_ws, size_t ws_size,
                              hipStream_t stream) {
  (void)in_sizes; (void)n_in; (void)out_size; (void)ws_size;

  const int*   node_types = (const int*)d_in[0];
  const float* pos  = (const float*)d_in[1];
  const float* rot  = (const float*)d_in[2];
  const float* siz  = (const float*)d_in[3];
  const int*   eidx = (const int*)d_in[4];
  const int*   etyp = (const int*)d_in[5];
  const unsigned char* nm = (const unsigned char*)d_in[6];
  const unsigned char* em = (const unsigned char*)d_in[7];
  const float* clip    = (const float*)d_in[8];
  const float* type_W  = (const float*)d_in[9];
  const float* type_b  = (const float*)d_in[10];
  const float* geom_W1 = (const float*)d_in[11];
  const float* geom_b1 = (const float*)d_in[12];
  const float* geom_W2 = (const float*)d_in[13];
  const float* geom_b2 = (const float*)d_in[14];
  const float* fuse_W  = (const float*)d_in[15];
  const float* fuse_b  = (const float*)d_in[16];
  const float* eembed  = (const float*)d_in[17];

  float* node_f32 = (float*)d_out;                         // (B,N,D) output 1
  float* gemb_out = (float*)d_out + (size_t)Bn * Nn * Dn;  // (B,D)   output 2

  // workspace carve (256B aligned)
  char* ws = (char*)d_ws;
  size_t off = 0;
  auto take = [&](size_t bytes) -> void* {
    void* p = ws + off;
    off = (off + bytes + 255) & ~(size_t)255;
    return p;
  };
  h16_t* nf16  = (h16_t*)take((size_t)Bn * Nn * Dn * 2);
  h16_t* tf16  = (h16_t*)take((size_t)Bn * Nn * Dn * 2);
  h16_t* gf16  = (h16_t*)take((size_t)Bn * Nn * Dn * 2);
  h16_t* ef16  = (h16_t*)take((size_t)Bn * En * Dn * 2);
  h16_t* msg16 = (h16_t*)take((size_t)Bn * En * Dn * 2);
  h16_t* hbuf  = (h16_t*)take((size_t)Bn * En * Dn * 2);
  float* agg   = (float*)take((size_t)Bn * Nn * Dn * 4);
  float* cnts  = (float*)take((size_t)Bn * Nn * 4);
  h16_t* typeWt  = (h16_t*)take((size_t)512 * Dn * 2);
  h16_t* geomW2t = (h16_t*)take((size_t)256 * Dn * 2);
  h16_t* fuseWt  = (h16_t*)take((size_t)512 * Dn * 2);
  h16_t *triW1t[3], *triW2t[3], *nodW1t[3], *nodW2t[3], *edgW1t[3], *edgW2t[3];
  for (int l = 0; l < 3; ++l) {
    triW1t[l] = (h16_t*)take((size_t)768 * Dn * 2);
    triW2t[l] = (h16_t*)take((size_t)256 * Dn * 2);
    nodW1t[l] = (h16_t*)take((size_t)256 * Dn * 2);
    nodW2t[l] = (h16_t*)take((size_t)256 * Dn * 2);
    edgW1t[l] = (h16_t*)take((size_t)256 * Dn * 2);
    edgW2t[l] = (h16_t*)take((size_t)256 * Dn * 2);
  }

  auto pack = [&](const float* W, h16_t* Wt, int K) {
    pack_k<<<(K * Dn + 255) / 256, 256, 0, stream>>>(W, Wt, K);
  };
  pack(type_W, typeWt, 512);
  pack(geom_W2, geomW2t, 256);
  pack(fuse_W, fuseWt, 512);
  for (int l = 0; l < 3; ++l) {
    const int LB = 18 + l * 14;
    pack((const float*)d_in[LB + 0], triW1t[l], 768);
    pack((const float*)d_in[LB + 2], triW2t[l], 256);
    pack((const float*)d_in[LB + 4], nodW1t[l], 256);
    pack((const float*)d_in[LB + 6], nodW2t[l], 256);
    pack((const float*)d_in[LB + 8], edgW1t[l], 256);
    pack((const float*)d_in[LB + 10], edgW2t[l], 256);
  }

  GemmP p;
  const int MN = Bn * Nn;   // 8192 node rows
  const int ME = Bn * En;   // 65536 edge rows

  // type_feat = clip_table[types] @ type_W + type_b   (gather fused into A-load)
  p = {}; p.A = clip; p.eidx = node_types; p.W = typeWt; p.bias = type_b;
  p.outF16 = tf16; p.M = MN; p.K = 512;
  gemm_k<LD_GATHER_F32, EP_F16><<<MN / 16, 128, 0, stream>>>(p);

  // geom MLP
  geom_hidden_k<<<MN, 256, 0, stream>>>(pos, rot, siz, geom_W1, geom_b1, hbuf);
  p = {}; p.A = hbuf; p.W = geomW2t; p.bias = geom_b2; p.outF16 = gf16;
  p.M = MN; p.K = 256;
  gemm_k<LD_F16, EP_F16><<<MN / 16, 128, 0, stream>>>(p);

  // fuse: node_feat = concat(type,geom) @ fuse_W + b, masked
  p = {}; p.A = tf16; p.A2 = gf16; p.W = fuseWt; p.bias = fuse_b; p.mask = nm;
  p.outF32 = node_f32; p.outF16 = nf16; p.M = MN; p.K = 512;
  gemm_k<LD_CONCAT2_F16, EP_FUSE><<<MN / 16, 128, 0, stream>>>(p);

  // edge_feat init
  edge_init_k<<<(Bn * En * Dn) / 256, 256, 0, stream>>>(etyp, em, eembed, ef16);

  for (int l = 0; l < 3; ++l) {
    const int LB = 18 + l * 14;
    const float* tri_b1  = (const float*)d_in[LB + 1];
    const float* tri_b2  = (const float*)d_in[LB + 3];
    const float* node_b1 = (const float*)d_in[LB + 5];
    const float* node_b2 = (const float*)d_in[LB + 7];
    const float* edge_b1 = (const float*)d_in[LB + 9];
    const float* edge_b2 = (const float*)d_in[LB + 11];
    const float* ln_g    = (const float*)d_in[LB + 12];
    const float* ln_b    = (const float*)d_in[LB + 13];

    (void)hipMemsetAsync(agg, 0, (size_t)Bn * Nn * Dn * 4, stream);
    (void)hipMemsetAsync(cnts, 0, (size_t)Bn * Nn * 4, stream);

    // triplet hidden = silu([src|edge|dst] @ tri_W1 + b1)   (gathers fused)
    p = {}; p.A = nf16; p.A2 = ef16; p.eidx = eidx; p.W = triW1t[l];
    p.bias = tri_b1; p.outF16 = hbuf; p.M = ME; p.K = 768;
    gemm_k<LD_TRIPLET, EP_SILU_F16><<<ME / 16, 128, 0, stream>>>(p);

    // msg = (hidden @ tri_W2 + b2) * em, fused scatter-add into agg
    p = {}; p.A = hbuf; p.W = triW2t[l]; p.bias = tri_b2; p.mask = em;
    p.eidx = eidx; p.outF16 = msg16; p.agg = agg; p.M = ME; p.K = 256;
    gemm_k<LD_F16, EP_MSG><<<ME / 16, 128, 0, stream>>>(p);

    counts_k<<<ME / 256, 256, 0, stream>>>(eidx, em, cnts);

    // edge update: ef += mlp2(msg)
    p = {}; p.A = msg16; p.W = edgW1t[l]; p.bias = edge_b1; p.outF16 = hbuf;
    p.M = ME; p.K = 256;
    gemm_k<LD_F16, EP_SILU_F16><<<ME / 16, 128, 0, stream>>>(p);
    p = {}; p.A = hbuf; p.W = edgW2t[l]; p.bias = edge_b2; p.outF16 = ef16;
    p.M = ME; p.K = 256;
    gemm_k<LD_F16, EP_EDGE_RES><<<ME / 16, 128, 0, stream>>>(p);

    // agg /= max(counts,1)
    aggnorm_k<<<(Bn * Nn * Dn) / 256, 256, 0, stream>>>(agg, cnts);

    // node update: node += mlp2(agg)*nm, then layernorm (fused)
    p = {}; p.A = agg; p.W = nodW1t[l]; p.bias = node_b1; p.outF16 = hbuf;
    p.M = MN; p.K = 256;
    gemm_k<LD_F32, EP_SILU_F16><<<MN / 16, 128, 0, stream>>>(p);
    p = {}; p.A = hbuf; p.W = nodW2t[l]; p.bias = node_b2; p.mask = nm;
    p.outF32 = node_f32; p.outF16 = nf16; p.lng = ln_g; p.lnb = ln_b;
    p.M = MN; p.K = 256;
    gemm_k<LD_F16, EP_NODE_LN><<<MN / 16, 128, 0, stream>>>(p);
  }

  // masked global mean
  gemb_k<<<Bn, 256, 0, stream>>>(node_f32, nm, gemb_out);
}
